// SessionGraph_10436770529471
// MI455X (gfx1250) — matrix operand order repaired and checked by reading, not compiled, not run on hardware
//
// SR-GNN session model forward for MI455X (gfx1250, wave32).
// - Final 512x99999 score GEMM: v_wmma_f32_16x16x32_f16, store-bound (205MB @ 23.3TB/s ~ 9us).
// - Batch-axis MHA (4x, 512x512x20 per seq slot): WMMA flash attention,
//   K/V staged to LDS as f16 (K row-major, V transposed), online softmax with
//   shfl_xor row reductions; P re-layout D-frag->A-frag via 1KB/wave LDS bounce.
#include <hip/hip_runtime.h>
#include <hip/hip_bf16.h>

#define BB 512
#define SS 50
#define HH 20
#define H3 60
#define NNODE 100000
#define NOUT 99999
#define TWO_S 100
#define BSH (BB*SS*HH)       // 512000
#define QSCALE 0.2236067977f // 1/sqrt(20)

typedef __attribute__((ext_vector_type(16))) _Float16 v16h;
typedef __attribute__((ext_vector_type(8)))  float    v8f;

union AFrag { v16h v; _Float16 h[16]; };
union CFrag { v8f v; float f[8]; };

__device__ __forceinline__ float dot20(const float* __restrict__ a,
                                       const float* __restrict__ w) {
  float s = 0.f;
#pragma unroll
  for (int k = 0; k < 20; ++k) s += a[k] * w[k];
  return s;
}

// ---------------- embedding gather ----------------
__global__ __launch_bounds__(256) void k_gather(
    const int* __restrict__ inputs, const int* __restrict__ category,
    const float* __restrict__ emb, const float* __restrict__ emb1,
    float* __restrict__ h0, float* __restrict__ cat) {
  int idx = blockIdx.x * 256 + threadIdx.x;
  if (idx >= BSH) return;
  int h = idx % HH; int bs = idx / HH;
  h0[idx]  = emb[(size_t)inputs[bs] * HH + h];
  cat[idx] = emb1[(size_t)category[bs] * HH + h];
}

// ---------------- GNN: edge linears ----------------
__global__ __launch_bounds__(256) void k_gnn_lin(
    const float* __restrict__ h0,
    const float* __restrict__ w_ein, const float* __restrict__ b_ein,
    const float* __restrict__ w_eout, const float* __restrict__ b_eout,
    float* __restrict__ hin, float* __restrict__ hout) {
  int idx = blockIdx.x * 256 + threadIdx.x;
  if (idx >= BSH) return;
  int h = idx % HH; int bs = idx / HH;
  const float* row = h0 + (size_t)bs * HH;
  hin[idx]  = dot20(row, w_ein  + h * HH) + b_ein[h];
  hout[idx] = dot20(row, w_eout + h * HH) + b_eout[h];
}

// ---------------- GNN: A-matrix batched matmul ----------------
__global__ __launch_bounds__(256) void k_gnn_amm(
    const float* __restrict__ A,
    const float* __restrict__ hin, const float* __restrict__ hout,
    const float* __restrict__ b_iah, const float* __restrict__ b_oah,
    float* __restrict__ iin, float* __restrict__ iout) {
  int idx = blockIdx.x * 256 + threadIdx.x;
  if (idx >= BSH) return;
  int h = idx % HH; int bs = idx / HH; int b = bs / SS; int i = bs % SS;
  const float* Ar = A + ((size_t)b * SS + i) * TWO_S;
  const float* hi = hin  + (size_t)b * SS * HH + h;
  const float* ho = hout + (size_t)b * SS * HH + h;
  float s0 = 0.f, s1 = 0.f;
#pragma unroll 10
  for (int j = 0; j < SS; ++j) {
    s0 += Ar[j]      * hi[j * HH];
    s1 += Ar[SS + j] * ho[j * HH];
  }
  iin[idx]  = s0 + b_iah[h];
  iout[idx] = s1 + b_oah[h];
}

// ---------------- GNN: GRU gate ----------------
__global__ __launch_bounds__(256) void k_gnn_gate(
    const float* __restrict__ h0,
    const float* __restrict__ iin, const float* __restrict__ iout,
    const float* __restrict__ w_ih, const float* __restrict__ b_ih,
    const float* __restrict__ w_hh, const float* __restrict__ b_hh,
    float* __restrict__ out) {
  int idx = blockIdx.x * 256 + threadIdx.x;
  if (idx >= BSH) return;
  int h = idx % HH; int bs = idx / HH;
  const float* ci = iin + (size_t)bs * HH;
  const float* co = iout + (size_t)bs * HH;
  const float* hr = h0 + (size_t)bs * HH;
  auto gi = [&](int c) {
    return dot20(ci, w_ih + c * 40) + dot20(co, w_ih + c * 40 + 20) + b_ih[c];
  };
  auto gh = [&](int c) { return dot20(hr, w_hh + c * HH) + b_hh[c]; };
  float r = 1.f / (1.f + __expf(-(gi(h) + gh(h))));
  float z = 1.f / (1.f + __expf(-(gi(HH + h) + gh(HH + h))));
  float n = tanhf(gi(2 * HH + h) + r * gh(2 * HH + h));
  float hv = hr[h];
  out[idx] = n + z * (hv - n);
}

// ---------------- last index ----------------
__global__ __launch_bounds__(256) void k_last(const int* __restrict__ mask,
                                              int* __restrict__ lastb) {
  int b = blockIdx.x * 256 + threadIdx.x;
  if (b >= BB) return;
  int s = 0;
#pragma unroll 10
  for (int j = 0; j < SS; ++j) s += mask[b * SS + j];
  lastb[b] = s - 1;
}

__global__ __launch_bounds__(256) void k_capture_ht(
    const float* __restrict__ xg, const float* __restrict__ cat,
    const int* __restrict__ lastb,
    float* __restrict__ htit, float* __restrict__ htcat) {
  int idx = blockIdx.x * 256 + threadIdx.x;
  if (idx >= BB * HH) return;
  int h = idx % HH; int b = idx / HH;
  int l = lastb[b];
  htit[idx]  = xg[((size_t)b * SS + l) * HH + h];
  htcat[idx] = cat[((size_t)b * SS + l) * HH + h];
}

// ---------------- MHA: in-projection (q scaled) ----------------
__global__ __launch_bounds__(256) void k_qkv(
    const float* __restrict__ x, const float* __restrict__ w,
    const float* __restrict__ bias, float* __restrict__ qkv) {
  int idx = blockIdx.x * 256 + threadIdx.x;
  if (idx >= BB * SS * H3) return;
  int c = idx % H3; int bs = idx / H3;
  float v = dot20(x + (size_t)bs * HH, w + c * HH) + bias[c];
  if (c < HH) v *= QSCALE;
  qkv[idx] = v;
}

// ---------------- MHA: WMMA flash attention over batch axis ----------------
// grid (n=50, 4); 8 waves/WG; wave w owns query tile ltile = blockIdx.y*8 + w.
// LDS: K chunk [128][32] f16, V^T chunk [32][128] f16, per-wave P bounce [16][32].
__global__ __launch_bounds__(256) void k_attn_wmma(
    const float* __restrict__ qkv, float* __restrict__ out) {
  __shared__ _Float16 Kh[128 * 32];
  __shared__ _Float16 VT[32 * 128];
  __shared__ _Float16 Pws[8][16 * 32];
  const int n = blockIdx.x;
  const int wave = threadIdx.x >> 5;
  const int lane = threadIdx.x & 31;
  const int khalf = lane >> 4;
  const int lmod = lane & 15;
  const int ltile = blockIdx.y * 8 + wave;      // 0..31

  // Q A-fragment: rows l = ltile*16 + lmod, K padded 20->32 (q already scaled)
  AFrag qf;
  const float* qrow = qkv + ((size_t)(ltile * 16 + lmod) * SS + n) * H3;
#pragma unroll
  for (int i = 0; i < 8; ++i) {
    int k = khalf * 16 + i;
    qf.h[i] = (k < HH) ? (_Float16)qrow[k] : (_Float16)0.f;
  }
#pragma unroll
  for (int i = 0; i < 8; ++i) {
    int k = khalf * 16 + 8 + i;
    qf.h[8 + i] = (k < HH) ? (_Float16)qrow[k] : (_Float16)0.f;
  }

  CFrag o0, o1;
  float rmax[8], rsum[8];
#pragma unroll
  for (int r = 0; r < 8; ++r) { o0.f[r] = 0.f; o1.f[r] = 0.f; rmax[r] = -1e30f; rsum[r] = 0.f; }

  for (int m0 = 0; m0 < BB; m0 += 128) {
    __syncthreads();
    // stage K (row-major) and V (transposed) as f16, zero-padded to K=32
    for (int t = threadIdx.x; t < 128 * 32; t += 256) {
      int mm = t >> 5, k = t & 31;
      const float* row = qkv + ((size_t)(m0 + mm) * SS + n) * H3;
      Kh[t] = (k < HH) ? (_Float16)row[HH + k] : (_Float16)0.f;
      VT[k * 128 + mm] = (k < HH) ? (_Float16)row[2 * HH + k] : (_Float16)0.f;
    }
    __syncthreads();
#pragma unroll
    for (int c0 = 0; c0 < 128; c0 += 32) {
      // B-fragments of K^T for keys c0..c0+31 (16 contiguous halves per lane)
      AFrag kf0, kf1;
#pragma unroll
      for (int i = 0; i < 16; ++i) kf0.h[i] = Kh[(c0 + lmod) * 32 + khalf * 16 + i];
#pragma unroll
      for (int i = 0; i < 16; ++i) kf1.h[i] = Kh[(c0 + 16 + lmod) * 32 + khalf * 16 + i];
      CFrag d0, d1;
#pragma unroll
      for (int r = 0; r < 8; ++r) { d0.f[r] = 0.f; d1.f[r] = 0.f; }
      d0.v = __builtin_amdgcn_wmma_f32_16x16x32_f16(false, qf.v, false, kf0.v,
                                                    (short)0, d0.v, false, false);
      d1.v = __builtin_amdgcn_wmma_f32_16x16x32_f16(false, qf.v, false, kf1.v,
                                                    (short)0, d1.v, false, false);
      // online softmax: D-frag row = r + 8*khalf, col = lmod (same map as O frags)
      float sc[8];
#pragma unroll
      for (int r = 0; r < 8; ++r) {
        float mx = fmaxf(d0.f[r], d1.f[r]);
#pragma unroll
        for (int msk = 1; msk < 16; msk <<= 1) mx = fmaxf(mx, __shfl_xor(mx, msk, 16));
        float nm = fmaxf(rmax[r], mx);
        sc[r] = __expf(rmax[r] - nm);
        rmax[r] = nm;
      }
#pragma unroll
      for (int r = 0; r < 8; ++r) {
        float p0 = __expf(d0.f[r] - rmax[r]);
        float p1 = __expf(d1.f[r] - rmax[r]);
        int mr = r + 8 * khalf;
        Pws[wave][mr * 32 + lmod]      = (_Float16)p0;
        Pws[wave][mr * 32 + 16 + lmod] = (_Float16)p1;
        float ps = p0 + p1;
#pragma unroll
        for (int msk = 1; msk < 16; msk <<= 1) ps += __shfl_xor(ps, msk, 16);
        rsum[r] = rsum[r] * sc[r] + ps;
        o0.f[r] *= sc[r];
        o1.f[r] *= sc[r];
      }
      // re-read P as A-fragment (per-wave region; same-wave LDS ops are in order)
      AFrag pf;
#pragma unroll
      for (int i = 0; i < 8; ++i) pf.h[i]     = Pws[wave][lmod * 32 + khalf * 16 + i];
#pragma unroll
      for (int i = 0; i < 8; ++i) pf.h[8 + i] = Pws[wave][lmod * 32 + khalf * 16 + 8 + i];
      // B-fragments of V (k = key index within chunk, col = output channel)
      AFrag vf0, vf1;
#pragma unroll
      for (int i = 0; i < 16; ++i) vf0.h[i] = VT[lmod * 128 + c0 + khalf * 16 + i];
#pragma unroll
      for (int i = 0; i < 16; ++i) vf1.h[i] = VT[(16 + lmod) * 128 + c0 + khalf * 16 + i];
      o0.v = __builtin_amdgcn_wmma_f32_16x16x32_f16(false, pf.v, false, vf0.v,
                                                    (short)0, o0.v, false, false);
      o1.v = __builtin_amdgcn_wmma_f32_16x16x32_f16(false, pf.v, false, vf1.v,
                                                    (short)0, o1.v, false, false);
    }
  }
  // write attnout[l][n][e] (e<20); frag0 cols 0..15, frag1 cols 16..19
#pragma unroll
  for (int r = 0; r < 8; ++r) {
    int l = ltile * 16 + r + 8 * khalf;
    float inv = 1.f / rsum[r];
    float* orow = out + ((size_t)l * SS + n) * HH;
    orow[lmod] = o0.f[r] * inv;
    int e1 = 16 + lmod;
    if (e1 < HH) orow[e1] = o1.f[r] * inv;
  }
}

// ---------------- MHA out-proj / residual MLP ----------------
__global__ __launch_bounds__(256) void k_proj(
    const float* __restrict__ attn, const float* __restrict__ w,
    const float* __restrict__ b, float* __restrict__ m) {
  int idx = blockIdx.x * 256 + threadIdx.x;
  if (idx >= BSH) return;
  int h = idx % HH; int bs = idx / HH;
  m[idx] = dot20(attn + (size_t)bs * HH, w + h * HH) + b[h];
}

__global__ __launch_bounds__(256) void k_res1(
    const float* __restrict__ m, const float* __restrict__ w1,
    const float* __restrict__ b1, float* __restrict__ t) {
  int idx = blockIdx.x * 256 + threadIdx.x;
  if (idx >= BSH) return;
  int j = idx % HH; int bs = idx / HH;
  float v = dot20(m + (size_t)bs * HH, w1 + j * HH) + b1[j];
  t[idx] = v > 0.f ? v : 0.f;
}

__global__ __launch_bounds__(256) void k_res2(
    const float* __restrict__ m, const float* __restrict__ t,
    const float* __restrict__ w2, const float* __restrict__ b2,
    float* __restrict__ x) {
  int idx = blockIdx.x * 256 + threadIdx.x;
  if (idx >= BSH) return;
  int h = idx % HH; int bs = idx / HH;
  x[idx] = m[idx] + dot20(t + (size_t)bs * HH, w2 + h * HH) + b2[h];
}

// ---------------- mix a = 0.52*x[last] + 0.48*ht ----------------
__global__ __launch_bounds__(256) void k_mix(
    const float* __restrict__ x, const float* __restrict__ ht,
    const int* __restrict__ lastb, float* __restrict__ a, int off) {
  int idx = blockIdx.x * 256 + threadIdx.x;
  if (idx >= BB * HH) return;
  int h = idx % HH; int b = idx / HH;
  int l = lastb[b];
  a[b * 40 + off + h] =
      0.52f * x[((size_t)b * SS + l) * HH + h] + 0.48f * ht[b * HH + h];
}

// ---------------- memory readout + gated head -> f16 final[512][32] ----------------
__global__ __launch_bounds__(256) void k_head(
    const float* __restrict__ a,
    const float* __restrict__ mem_M, const float* __restrict__ mem_Wa,
    const float* __restrict__ mem_fc,
    const float* __restrict__ wc, const float* __restrict__ bc,
    const float* __restrict__ wp, const float* __restrict__ bp,
    _Float16* __restrict__ finalh) {
  int b = blockIdx.x * 256 + threadIdx.x;
  if (b >= BB) return;
  float av[40];
#pragma unroll
  for (int k = 0; k < 40; ++k) av[k] = a[b * 40 + k];
  float q[HH];
#pragma unroll
  for (int m = 0; m < HH; ++m) {
    float s = 0.f;
#pragma unroll
    for (int k = 0; k < 40; ++k) s += av[k] * mem_Wa[k * HH + m];
    q[m] = s;
  }
  float lg[10]; float mx = -1e30f;
#pragma unroll
  for (int c = 0; c < 10; ++c) {
    float s = 0.f;
#pragma unroll
    for (int m = 0; m < HH; ++m) s += q[m] * mem_M[c * HH + m];
    lg[c] = s; mx = s > mx ? s : mx;
  }
  float se = 0.f;
#pragma unroll
  for (int c = 0; c < 10; ++c) { lg[c] = __expf(lg[c] - mx); se += lg[c]; }
  float inv = 1.f / se;
  float rep0[HH];
#pragma unroll
  for (int m = 0; m < HH; ++m) {
    float s = 0.f;
#pragma unroll
    for (int c = 0; c < 10; ++c) s += lg[c] * inv * mem_M[c * HH + m];
    rep0[m] = s;
  }
  float car[60];
#pragma unroll
  for (int k = 0; k < 40; ++k) car[k] = av[k];
#pragma unroll
  for (int m2 = 0; m2 < HH; ++m2) {
    float s = 0.f;
#pragma unroll
    for (int m = 0; m < HH; ++m) s += rep0[m] * mem_fc[m * HH + m2];
    car[40 + m2] = tanhf(s);
  }
#pragma unroll
  for (int h = 0; h < HH; ++h) {
    float s1 = bc[h], s2 = bp[h];
#pragma unroll
    for (int k = 0; k < 60; ++k) { s1 += car[k] * wc[h * 60 + k]; s2 += car[k] * wp[h * 60 + k]; }
    float f = (1.f / (1.f + __expf(-s1))) * s2;
    finalh[b * 32 + h] = (_Float16)f;
  }
#pragma unroll
  for (int k = HH; k < 32; ++k) finalh[b * 32 + k] = (_Float16)0.f;
}

// ---------------- emb -> f16, K padded 20->32 ----------------
__global__ __launch_bounds__(256) void k_embh(const float* __restrict__ emb,
                                              _Float16* __restrict__ embh) {
  int idx = blockIdx.x * 256 + threadIdx.x;
  if (idx >= NNODE * 32) return;
  int k = idx & 31; int node = idx >> 5;
  embh[idx] = (k < HH) ? (_Float16)emb[(size_t)node * HH + k] : (_Float16)0.f;
}

// ---------------- scores = final @ emb[1:].T via WMMA f16->f32 ----------------
__global__ __launch_bounds__(32) void k_scores(
    const _Float16* __restrict__ Ah /*[512][32]*/,
    const _Float16* __restrict__ Bh /*[100000][32]*/,
    float* __restrict__ out /*[512][99999]*/) {
  int lane = threadIdx.x;
  int ntile = blockIdx.x;   // 0..6249
  int mtile = blockIdx.y;   // 0..31
  int khalf = lane >> 4;
  int mrow = mtile * 16 + (lane & 15);
  AFrag a;
  const _Float16* arow = Ah + (size_t)mrow * 32;
#pragma unroll
  for (int i = 0; i < 8; ++i) a.h[i]     = arow[khalf * 16 + i];
#pragma unroll
  for (int i = 0; i < 8; ++i) a.h[8 + i] = arow[khalf * 16 + 8 + i];
  int j = ntile * 16 + (lane & 15);
  int jc = j < NOUT ? j : (NOUT - 1);
  const _Float16* brow = Bh + (size_t)(1 + jc) * 32 + khalf * 16;
  AFrag bf;
#pragma unroll
  for (int i = 0; i < 16; ++i) bf.h[i] = brow[i];
  CFrag c;
#pragma unroll
  for (int r = 0; r < 8; ++r) c.f[r] = 0.f;
  c.v = __builtin_amdgcn_wmma_f32_16x16x32_f16(
      false, a.v, false, bf.v, (short)0, c.v, false, false);
  if (j < NOUT) {
#pragma unroll
    for (int r = 0; r < 8; ++r) {
      int m = mtile * 16 + r + 8 * khalf;
      out[(size_t)m * NOUT + j] = c.f[r];
    }
  }
}

extern "C" void kernel_launch(void* const* d_in, const int* in_sizes, int n_in,
                              void* d_out, int out_size, void* d_ws, size_t ws_size,
                              hipStream_t stream) {
  const int*   inputs   = (const int*)d_in[0];
  const int*   category = (const int*)d_in[1];
  const float* A        = (const float*)d_in[2];
  const int*   mask     = (const int*)d_in[3];
  const float* emb      = (const float*)d_in[4];
  const float* emb1     = (const float*)d_in[5];
  const float* w_ih     = (const float*)d_in[6];
  const float* w_hh     = (const float*)d_in[7];
  const float* b_ih     = (const float*)d_in[8];
  const float* b_hh     = (const float*)d_in[9];
  const float* b_iah    = (const float*)d_in[10];
  const float* b_oah    = (const float*)d_in[11];
  const float* w_ein    = (const float*)d_in[12];
  const float* b_ein    = (const float*)d_in[13];
  const float* w_eout   = (const float*)d_in[14];
  const float* b_eout   = (const float*)d_in[15];
  const float* mha_in_w  = (const float*)d_in[16];
  const float* mha_in_b  = (const float*)d_in[17];
  const float* mha_out_w = (const float*)d_in[18];
  const float* mha_out_b = (const float*)d_in[19];
  const float* mha1_in_w  = (const float*)d_in[20];
  const float* mha1_in_b  = (const float*)d_in[21];
  const float* mha1_out_w = (const float*)d_in[22];
  const float* mha1_out_b = (const float*)d_in[23];
  const float* rn_w1 = (const float*)d_in[24];
  const float* rn_b1 = (const float*)d_in[25];
  const float* rn_w2 = (const float*)d_in[26];
  const float* rn_b2 = (const float*)d_in[27];
  const float* rn1_w1 = (const float*)d_in[28];
  const float* rn1_b1 = (const float*)d_in[29];
  const float* rn1_w2 = (const float*)d_in[30];
  const float* rn1_b2 = (const float*)d_in[31];
  const float* mem_M  = (const float*)d_in[32];
  const float* mem_Wa = (const float*)d_in[33];
  const float* mem_fc = (const float*)d_in[34];
  const float* wc = (const float*)d_in[35];
  const float* bc = (const float*)d_in[36];
  const float* wp = (const float*)d_in[37];
  const float* bp = (const float*)d_in[38];

  float* ws = (float*)d_ws;
  // layout (floats); total footprint ~23 MB
  float* h0   = ws;                 // [512000]  (reused as mbuf after GNN)
  float* cat  = ws + 512000;        // [512000]
  float* bufA = ws + 1024000;       // GNN hin   (later: qkv region start)
  float* bufB = ws + 1536000;       // GNN hout
  float* bufC = ws + 2048000;       // GNN inp_in
  float* bufD = ws + 2560000;       // GNN inp_out (later: attnout)
  float* xbuf = ws + 3072000;       // chain state
  float* qkvb = ws + 1024000;       // [1536000] aliases bufA..bufC
  float* attb = ws + 2560000;       // [512000]  aliases bufD
  float* mbuf = ws;                 // aliases h0 (free after GNN gate)
  float* tbuf = ws + 3584000;       // [512000]
  int*   lastb = (int*)(ws + 4096000);
  float* htit  = ws + 4097024;
  float* htcat = ws + 4107264;
  float* abuf  = ws + 4117504;      // [512*40]
  _Float16* finalh = (_Float16*)(ws + 4140000); // 512*32 halves
  _Float16* embh   = (_Float16*)(ws + 4150000); // 100000*32 halves
  (void)in_sizes; (void)n_in; (void)out_size; (void)ws_size;

  const int G_BSH = (BSH + 255) / 256;            // 2000
  const int G_QKV = (BB * SS * H3 + 255) / 256;   // 6000
  const int G_BH  = (BB * HH + 255) / 256;        // 40

  // 1. embeddings
  k_gather<<<G_BSH, 256, 0, stream>>>(inputs, category, emb, emb1, h0, cat);
  // 2-4. GNN cell (STEP=1)
  k_gnn_lin<<<G_BSH, 256, 0, stream>>>(h0, w_ein, b_ein, w_eout, b_eout, bufA, bufB);
  k_gnn_amm<<<G_BSH, 256, 0, stream>>>(A, bufA, bufB, b_iah, b_oah, bufC, bufD);
  k_gnn_gate<<<G_BSH, 256, 0, stream>>>(h0, bufC, bufD, w_ih, b_ih, w_hh, b_hh, xbuf);
  // 5-6. last positions, ht captures
  k_last<<<2, 256, 0, stream>>>(mask, lastb);
  k_capture_ht<<<G_BH, 256, 0, stream>>>(xbuf, cat, lastb, htit, htcat);
  // 7. item chain: 2 x (MHA + residual), state in xbuf
  for (int kb = 0; kb < 2; ++kb) {
    k_qkv<<<G_QKV, 256, 0, stream>>>(xbuf, mha_in_w, mha_in_b, qkvb);
    k_attn_wmma<<<dim3(SS, 4), 256, 0, stream>>>(qkvb, attb);
    k_proj<<<G_BSH, 256, 0, stream>>>(attb, mha_out_w, mha_out_b, mbuf);
    k_res1<<<G_BSH, 256, 0, stream>>>(mbuf, rn_w1, rn_b1, tbuf);
    k_res2<<<G_BSH, 256, 0, stream>>>(mbuf, tbuf, rn_w2, rn_b2, xbuf);
  }
  k_mix<<<G_BH, 256, 0, stream>>>(xbuf, htit, lastb, abuf, 0);
  // 8. cat chain (block 0 reads cat_h, then xbuf)
  for (int kb = 0; kb < 2; ++kb) {
    const float* src = (kb == 0) ? cat : xbuf;
    k_qkv<<<G_QKV, 256, 0, stream>>>(src, mha1_in_w, mha1_in_b, qkvb);
    k_attn_wmma<<<dim3(SS, 4), 256, 0, stream>>>(qkvb, attb);
    k_proj<<<G_BSH, 256, 0, stream>>>(attb, mha1_out_w, mha1_out_b, mbuf);
    k_res1<<<G_BSH, 256, 0, stream>>>(mbuf, rn1_w1, rn1_b1, tbuf);
    k_res2<<<G_BSH, 256, 0, stream>>>(mbuf, tbuf, rn1_w2, rn1_b2, xbuf);
  }
  k_mix<<<G_BH, 256, 0, stream>>>(xbuf, htcat, lastb, abuf, HH);
  // 9. memory readout + gated head -> f16 final (padded K=32)
  k_head<<<2, 256, 0, stream>>>(abuf, mem_M, mem_Wa, mem_fc, wc, bc, wp, bp, finalh);
  // 10. emb table -> f16 (padded K=32)
  k_embh<<<(NNODE * 32 + 255) / 256, 256, 0, stream>>>(emb, embh);
  // 11. scores: [512,20] x [20,99999] via v_wmma_f32_16x16x32_f16
  k_scores<<<dim3((NOUT + 15) / 16, BB / 16), 32, 0, stream>>>(finalh, embh, (float*)d_out);
}
